// ITDA_71743133712764
// MI455X (gfx1250) — compile-verified, matrix operand order checked by reading, not run on hardware
//
#include <hip/hip_runtime.h>
#include <hip/hip_bf16.h>

// ---------------------------------------------------------------------------
// ITDA forward on MI455X (gfx1250, wave32, WMMA).
//   x,y: [1024,1024] f32   xs,ys: [8192,1024] f32 (xs unit-norm)  mean_y: [1024]
//   grad-pursuit (16 steps) -> topk(16) -> decode -> FVU losses.
// Dense work = residual @ xs^T per step via v_wmma_f32_16x16x32_bf16
// (A fragments from LDS, B fragments streamed from L2-resident packed dict).
// All sparse state (<=16 atoms/row) kept exactly in fp32.
// ---------------------------------------------------------------------------

typedef __attribute__((ext_vector_type(16))) __bf16 v16bf;
typedef __attribute__((ext_vector_type(8)))  float  v8f;

#define B_   1024
#define V_   1024
#define F_   8192
#define K_   16              // target_l0
#define NT_  (F_ / 16)       // 512 atom tiles (N)
#define KT_  (V_ / 32)       // 32  k tiles    (K)
#define FRAG 512             // bf16 elems per 16x32 fragment (32 lanes * 16)
#define THREADS 512
#define WAVES   (THREADS / 32)   // 16 waves; wave r owns batch row r
#define EPS_ 1e-3f

// ---- LDS layout (dynamic) --------------------------------------------------
#define OFF_RES     0                       // float[16][1024]        65536 B
#define OFF_RESB    65536                   // __bf16[KT_*FRAG]       32768 B
#define OFF_REDV    (65536 + 32768)         // float[WAVES*16]         1024 B
#define OFF_REDI    (OFF_REDV + 1024)       // int  [WAVES*16]         1024 B
#define OFF_SELCNT  (OFF_REDI + 1024)       // int[16]                   64 B
#define OFF_SELIDX  (OFF_SELCNT + 64)       // int[16][16]             1024 B
#define OFF_SELW    (OFF_SELIDX + 1024)     // float[16][16]           1024 B
#define OFF_GRAD    (OFF_SELW  + 1024)      // float[16][16]           1024 B
#define SMEM_BYTES  (OFF_GRAD  + 1024)      // ~101 KB

// ---------------------------------------------------------------------------
// Repack xs (f32 row-major) -> bf16 WMMA B-fragments.
// Fragment (nt,kk) holds atoms [nt*16,nt*16+16) x K [kk*32,kk*32+32).
// Lane l<16  : column N=l,  halves = K {0..7, 16..23}
// Lane 16+l  : column N=l,  halves = K {8..15, 24..31}
// Each lane's 16 bf16 are contiguous (32B) -> one wide load per lane.
// ---------------------------------------------------------------------------
__global__ void k_convert(const float* __restrict__ xs, __bf16* __restrict__ xsb) {
  int gid = blockIdx.x * blockDim.x + threadIdx.x;   // one thread per (frag,lane)
  if (gid >= NT_ * KT_ * 32) return;
  int lane = gid & 31;
  int f    = gid >> 5;
  int kk   = f & (KT_ - 1);
  int nt   = f >> 5;                                  // KT_ == 32
  int h = lane >> 4, n = lane & 15;
  __bf16*      dst = xsb + (size_t)f * FRAG + lane * 16;
  const float* src = xs + (size_t)(nt * 16 + n) * V_ + kk * 32;
#pragma unroll
  for (int i = 0; i < 16; ++i) {
    int klocal = (i < 8) ? (i + 8 * h) : ((i - 8) + 16 + 8 * h);
    dst[i] = (__bf16)src[klocal];
  }
}

// ---- total_var = mean_b sum_d (y - mean_y)^2, deterministic 2-stage --------
__global__ void k_totvar(const float* __restrict__ y, const float* __restrict__ mean_y,
                         float* __restrict__ partial) {
  __shared__ float sm[256];
  int r = blockIdx.x, t = threadIdx.x;
  float acc = 0.f;
  for (int d = t; d < V_; d += 256) { float v = y[(size_t)r * V_ + d] - mean_y[d]; acc += v * v; }
  sm[t] = acc; __syncthreads();
  for (int s = 128; s > 0; s >>= 1) { if (t < s) sm[t] += sm[t + s]; __syncthreads(); }
  if (t == 0) partial[r] = sm[0];
}

__global__ void k_finalize(const float* __restrict__ partial, float* __restrict__ tv) {
  __shared__ float sm[256];
  int t = threadIdx.x; float a = 0.f;
  for (int i = t; i < B_; i += 256) a += partial[i];
  sm[t] = a; __syncthreads();
  for (int s = 128; s > 0; s >>= 1) { if (t < s) sm[t] += sm[t + s]; __syncthreads(); }
  if (t == 0) *tv = sm[0] / (float)B_;
}

// ---------------------------------------------------------------------------
// Main pursuit kernel: one block = 16 batch rows, 512 threads = 16 waves.
// ---------------------------------------------------------------------------
__global__ __launch_bounds__(THREADS)
void k_pursuit(const float* __restrict__ x, const float* __restrict__ y,
               const float* __restrict__ xs, const float* __restrict__ ys,
               const __bf16* __restrict__ xsb, const float* __restrict__ tv,
               float* __restrict__ out_w, int* __restrict__ out_i,
               float* __restrict__ out_xrec, float* __restrict__ out_yrec,
               float* __restrict__ out_loss) {
  extern __shared__ char smem[];
  float*  res    = (float*)(smem + OFF_RES);      // [16][1024] fp32 residual
  __bf16* resb   = (__bf16*)(smem + OFF_RESB);    // bf16 A-fragment mirror
  float*  redv   = (float*)(smem + OFF_REDV);
  int*    redi   = (int*)  (smem + OFF_REDI);
  int*    selcnt = (int*)  (smem + OFF_SELCNT);
  int*    selidx = (int*)  (smem + OFF_SELIDX);
  float*  selw   = (float*)(smem + OFF_SELW);
  float*  grads  = (float*)(smem + OFF_GRAD);

  const int tid  = threadIdx.x;
  const int lane = tid & 31;
  const int wave = tid >> 5;        // == batch row owned by this wave
  const int r0   = blockIdx.x * 16;

  // residual := x rows; selected sets empty
  for (int e = tid; e < 16 * V_; e += THREADS)
    res[e] = x[(size_t)(r0 + (e >> 10)) * V_ + (e & (V_ - 1))];
  if (tid < 16) selcnt[tid] = 0;
  __syncthreads();

  auto refresh_resb = [&]() {   // fp32 residual -> bf16 A fragments (WMMA layout)
    for (int e = tid; e < KT_ * FRAG; e += THREADS) {
      int kk = e >> 9, rem = e & 511, ln = rem >> 4, i = rem & 15;
      int h = ln >> 4, m = ln & 15;
      int klocal = (i < 8) ? (i + 8 * h) : ((i - 8) + 16 + 8 * h);
      resb[e] = (__bf16)res[m * V_ + kk * 32 + klocal];
    }
  };
  refresh_resb();
  __syncthreads();

  for (int step = 0; step < K_; ++step) {
    // ---- inner = residual @ xs^T (bf16 WMMA, 2 atom-tiles per pass) -------
    // Per-lane tracking uses strict '>' only: atoms ascend per lane, so
    // keep-old-on-equal == lowest-index tie-break; idx compares happen only
    // in the log2 reductions below.
    float bv[8]; int bi[8];
#pragma unroll
    for (int g = 0; g < 8; ++g) { bv[g] = -1e30f; bi[g] = 0; }

    for (int nt = wave * 2; nt < NT_; nt += WAVES * 2) {
      v8f acc0 = {}, acc1 = {};
      const __bf16* bbase0 = xsb + ((size_t)nt * KT_) * FRAG + lane * 16;
      const __bf16* bbase1 = bbase0 + (size_t)KT_ * FRAG;
#pragma unroll 2
      for (int kk = 0; kk < KT_; ++kk) {
        v16bf a  = *(const v16bf*)(resb + kk * FRAG + lane * 16);   // LDS (shared A)
        v16bf b0 = *(const v16bf*)(bbase0 + kk * FRAG);             // L2-resident dict
        v16bf b1 = *(const v16bf*)(bbase1 + kk * FRAG);
        acc0 = __builtin_amdgcn_wmma_f32_16x16x32_bf16(
            false, a, false, b0, (short)0, acc0, false, false);
        acc1 = __builtin_amdgcn_wmma_f32_16x16x32_bf16(
            false, a, false, b1, (short)0, acc1, false, false);
      }
      int atom0 = nt * 16 + (lane & 15);   // C layout: N = lane & 15
      int atom1 = atom0 + 16;
#pragma unroll
      for (int g = 0; g < 8; ++g) {
        if (acc0[g] > bv[g]) { bv[g] = acc0[g]; bi[g] = atom0; }
        if (acc1[g] > bv[g]) { bv[g] = acc1[g]; bi[g] = atom1; }
      }
    }
    // argmax across the 16 columns held in each 16-lane half (rows g / g+8)
#pragma unroll
    for (int g = 0; g < 8; ++g) {
#pragma unroll
      for (int off = 8; off >= 1; off >>= 1) {
        float ov = __shfl_xor(bv[g], off, 32);
        int   oi = __shfl_xor(bi[g], off, 32);
        if (ov > bv[g] || (ov == bv[g] && oi < bi[g])) { bv[g] = ov; bi[g] = oi; }
      }
    }
    if ((lane & 15) == 0) {
      int half = lane >> 4;
#pragma unroll
      for (int g = 0; g < 8; ++g) {
        redv[wave * 16 + half * 8 + g] = bv[g];
        redi[wave * 16 + half * 8 + g] = bi[g];
      }
    }
    __syncthreads();

    // ---- per-row: cross-wave argmax, prune zero weights, append new atom --
    if (tid < 16) {
      float mv = -1e30f; int mi = 0;
      for (int w = 0; w < WAVES; ++w) {
        float v = redv[w * 16 + tid]; int i = redi[w * 16 + tid];
        if (v > mv || (v == mv && i < mi)) { mv = v; mi = i; }
      }
      int cnt = selcnt[tid], nc = 0;
      for (int j = 0; j < cnt; ++j)                    // selected = {w != 0}
        if (selw[tid * 16 + j] != 0.f) {
          selidx[tid * 16 + nc] = selidx[tid * 16 + j];
          selw[tid * 16 + nc]   = selw[tid * 16 + j];
          ++nc;
        }
      bool present = false;
      for (int j = 0; j < nc; ++j) present |= (selidx[tid * 16 + j] == mi);
      if (!present && nc < 16) { selidx[tid * 16 + nc] = mi; selw[tid * 16 + nc] = 0.f; ++nc; }
      selcnt[tid] = nc;
    }
    __syncthreads();

    // ---- grad_j = <residual_r, xs[idx_j]> : wave r owns row r -------------
    // strided d = lane + 32*i -> conflict-free LDS banks, coalesced global.
    {
      const float* rr = res + wave * V_;
      int cnt = selcnt[wave];
      for (int j = 0; j < cnt; ++j) {
        const float* atom = xs + (size_t)selidx[wave * 16 + j] * V_;
        float acc = 0.f;
#pragma unroll 8
        for (int i = 0; i < 32; ++i) { int d = lane + 32 * i; acc += rr[d] * atom[d]; }
#pragma unroll
        for (int off = 16; off >= 1; off >>= 1) acc += __shfl_xor(acc, off, 32);
        if (lane == 0) grads[wave * 16 + j] = acc;
      }
    }
    __syncthreads();

    // ---- c = grad@D (active set); step = <c,res>/max(<c,c>,eps); ReLU -----
    {
      float cvec[32];
#pragma unroll
      for (int i = 0; i < 32; ++i) cvec[i] = 0.f;
      int cnt = selcnt[wave];
      for (int j = 0; j < cnt; ++j) {
        float g = grads[wave * 16 + j];
        const float* atom = xs + (size_t)selidx[wave * 16 + j] * V_;
#pragma unroll 8
        for (int i = 0; i < 32; ++i) cvec[i] += g * atom[lane + 32 * i];
      }
      float cr = 0.f, cc = 0.f;
      const float* rr = res + wave * V_;
#pragma unroll
      for (int i = 0; i < 32; ++i) { float c = cvec[i]; cr += c * rr[lane + 32 * i]; cc += c * c; }
#pragma unroll
      for (int off = 16; off >= 1; off >>= 1) {
        cr += __shfl_xor(cr, off, 32);
        cc += __shfl_xor(cc, off, 32);
      }
      if (lane == 0) {
        float ss = cr / fmaxf(cc, EPS_);
        for (int j = 0; j < cnt; ++j)
          selw[wave * 16 + j] = fmaxf(selw[wave * 16 + j] + ss * grads[wave * 16 + j], 0.f);
      }
    }
    __syncthreads();

    // ---- exact residual rebuild: res = x - sum_j w_j * xs[idx_j] ----------
    {
      const float* xr = x + (size_t)(r0 + wave) * V_;
      float acc2[32];
#pragma unroll 8
      for (int i = 0; i < 32; ++i) acc2[i] = xr[lane + 32 * i];
      int cnt = selcnt[wave];
      for (int j = 0; j < cnt; ++j) {
        float w = selw[wave * 16 + j];
        const float* atom = xs + (size_t)selidx[wave * 16 + j] * V_;
#pragma unroll 8
        for (int i = 0; i < 32; ++i) acc2[i] -= w * atom[lane + 32 * i];
      }
      float* rr = res + wave * V_;
#pragma unroll
      for (int i = 0; i < 32; ++i) rr[lane + 32 * i] = acc2[i];
    }
    __syncthreads();
    refresh_resb();
    __syncthreads();
  }

  // ---- top-k (sort nonzero weights desc, tie -> lower idx; pad zeros) -----
  if (tid < 16) {
    int r = tid;
    float wv[16]; int wi[16];
    int cnt = selcnt[r], m = 0;
    for (int j = 0; j < cnt; ++j)
      if (selw[r * 16 + j] > 0.f) { wv[m] = selw[r * 16 + j]; wi[m] = selidx[r * 16 + j]; ++m; }
    for (int a = 0; a < m; ++a) {
      int best = a;
      for (int b = a + 1; b < m; ++b)
        if (wv[b] > wv[best] || (wv[b] == wv[best] && wi[b] < wi[best])) best = b;
      float tw = wv[a]; wv[a] = wv[best]; wv[best] = tw;
      int   ti = wi[a]; wi[a] = wi[best]; wi[best] = ti;
    }
    int f = 0;                                   // top_k tie-break: smallest unused idx
    for (int a = m; a < 16; ++a) {
      for (;; ++f) { bool used = false; for (int b = 0; b < m; ++b) used |= (wi[b] == f); if (!used) break; }
      wv[a] = 0.f; wi[a] = f; ++f;
    }
    for (int a = 0; a < 16; ++a) {
      out_w[(size_t)(r0 + r) * 16 + a] = wv[a];
      out_i[(size_t)(r0 + r) * 16 + a] = wi[a];
    }
  }
  __syncthreads();

  // ---- decode + losses:  x_rec = x - res (free);  y_rec from ys gathers ---
  {
    const float* xr = x + (size_t)(r0 + wave) * V_;
    const float* rr = res + wave * V_;
    float* xo = out_xrec + (size_t)(r0 + wave) * V_;
#pragma unroll 8
    for (int i = 0; i < 32; ++i) { int d = lane + 32 * i; xo[d] = xr[d] - rr[d]; }

    float yrec[32];
#pragma unroll
    for (int i = 0; i < 32; ++i) yrec[i] = 0.f;
    int cnt = selcnt[wave];
    for (int j = 0; j < cnt; ++j) {
      float w = selw[wave * 16 + j];
      const float* atom = ys + (size_t)selidx[wave * 16 + j] * V_;
#pragma unroll 8
      for (int i = 0; i < 32; ++i) yrec[i] += w * atom[lane + 32 * i];
    }
    float* yo = out_yrec + (size_t)(r0 + wave) * V_;
    const float* yr = y + (size_t)(r0 + wave) * V_;
    float l2 = 0.f;
#pragma unroll
    for (int i = 0; i < 32; ++i) {
      int d = lane + 32 * i;
      yo[d] = yrec[i];
      float e = yr[d] - yrec[i];
      l2 += e * e;
    }
#pragma unroll
    for (int off = 16; off >= 1; off >>= 1) l2 += __shfl_xor(l2, off, 32);
    if (lane == 0) out_loss[r0 + wave] = l2 / tv[0];
  }
}

// ---------------------------------------------------------------------------
extern "C" void kernel_launch(void* const* d_in, const int* in_sizes, int n_in,
                              void* d_out, int out_size, void* d_ws, size_t ws_size,
                              hipStream_t stream) {
  const float* x      = (const float*)d_in[0];   // [1024,1024]
  const float* y      = (const float*)d_in[1];   // [1024,1024]
  const float* xs     = (const float*)d_in[2];   // [8192,1024]
  const float* ys     = (const float*)d_in[3];   // [8192,1024]
  const float* mean_y = (const float*)d_in[4];   // [1024]
  (void)in_sizes; (void)n_in; (void)out_size;    // target_l0 hardcoded = 16

  // workspace: [ xsb bf16 16MB | tv partials 4KB | tv scalar ]
  __bf16* xsb      = (__bf16*)d_ws;
  const size_t XSB = (size_t)F_ * V_ * sizeof(__bf16);
  float*  partials = (float*)((char*)d_ws + XSB);
  float*  tv       = (float*)((char*)d_ws + XSB + 4096);
  (void)ws_size;

  // output: weights[B,16] | indices[B,16] | x_rec[B,V] | y_rec[B,V] | losses[B]
  float* out_w    = (float*)d_out;
  int*   out_i    = (int*)(out_w + (size_t)B_ * 16);
  float* out_xrec = (float*)(out_w + 2 * (size_t)B_ * 16);
  float* out_yrec = out_xrec + (size_t)B_ * V_;
  float* out_loss = out_yrec + (size_t)B_ * V_;

  {
    int total = NT_ * KT_ * 32;
    k_convert<<<(total + 255) / 256, 256, 0, stream>>>(xs, xsb);
  }
  k_totvar<<<B_, 256, 0, stream>>>(y, mean_y, partials);
  k_finalize<<<1, 256, 0, stream>>>(partials, tv);
  k_pursuit<<<B_ / 16, THREADS, SMEM_BYTES, stream>>>(
      x, y, xs, ys, xsb, tv, out_w, out_i, out_xrec, out_yrec, out_loss);
}
